// Block_79396765433942
// MI455X (gfx1250) — compile-verified, matrix operand order checked by reading, not run on hardware
//
#include <hip/hip_runtime.h>
#include <math.h>

typedef __bf16 bf16;
typedef __attribute__((ext_vector_type(16))) __bf16 v16bf;
typedef __attribute__((ext_vector_type(8)))  float  v8f;
typedef int v4i __attribute__((vector_size(16)));

static constexpr int  CD  = 24, CH = 96, CW = 192;
static constexpr int  CHW = CH * CW;                  // 18432
static constexpr long SP  = (long)CD * CH * CW;       // 442368
static constexpr int  DIM = 64, HIDC = 256;
static constexpr int  NB   = 128;                     // spatial cols per GEMM block
static constexpr int  NBLK = (int)(SP / NB);          // 3456

__device__ __forceinline__ float gelu_erf(float x) {
  return 0.5f * x * (1.0f + erff(x * 0.70710678118654752440f));
}

// ---------------------------------------------------------------------------
// CDNA5 async global->LDS path (ASYNCcnt-tracked), guarded per-toolchain.
// Builtin prototype (from probe diagnostic): (v4i as1*, v4i as3*, imm, imm).
// ---------------------------------------------------------------------------
#if defined(__HIP_DEVICE_COMPILE__) && __has_builtin(__builtin_amdgcn_global_load_async_to_lds_b128)
#define HAVE_ASYNC_LDS 1
#endif

typedef __attribute__((address_space(1))) v4i gv4i;
typedef __attribute__((address_space(3))) v4i lv4i;

__device__ __forceinline__ gv4i* to_gm128(const void* p) {
  // generic global pointer is numerically identical to AS1
  return (gv4i*)(unsigned long long)p;
}
__device__ __forceinline__ lv4i* to_lds128(void* p) {
  // generic LDS pointer: low 32 bits are the LDS byte offset (flat aperture rules)
  return (lv4i*)(unsigned int)(unsigned long long)p;
}

__device__ __forceinline__ void wait_async0() {
#if defined(__HIP_DEVICE_COMPILE__) && __has_builtin(__builtin_amdgcn_s_wait_asynccnt)
  __builtin_amdgcn_s_wait_asynccnt(0);
#elif defined(__HIP_DEVICE_COMPILE__)
  asm volatile("s_wait_asynccnt 0x0" ::: "memory");
#endif
}

// ---------------------------------------------------------------------------
// WMMA fragment helpers (wave32, 16x16x32 bf16, f32 accum)
// A (16x32): lane L holds row M = L&15; K-base kb = (L&16)?8:0.
//   elem i<8 -> K = kb+i ; elem i>=8 -> K = kb+16+(i-8)
// B is staged K-transposed in LDS as blds[n*ldk + k], same per-lane K map,
// with N = L&15.
// C/D: VGPR r -> M = r + ((L&16)?8:0), N = L&15.
// ---------------------------------------------------------------------------
__device__ __forceinline__ v16bf load_a_w(const float* __restrict__ w,
                                          int m, int K, int k0, int lane) {
  const int kb = (lane & 16) ? 8 : 0;
  const float* p = w + (long)m * K + k0 + kb;
  v16bf a;
#pragma unroll
  for (int i = 0; i < 8; ++i) { a[i] = (bf16)p[i]; a[i + 8] = (bf16)p[i + 16]; }
  return a;
}

__device__ __forceinline__ v16bf load_b_lds(const bf16* blds, int n, int ldk,
                                            int k0, int lane) {
  const int kb = (lane & 16) ? 8 : 0;
  const bf16* p = blds + (long)n * ldk + k0 + kb;
  v16bf b;
#pragma unroll
  for (int i = 0; i < 8; ++i) { b[i] = p[i]; b[i + 8] = p[i + 16]; }
  return b;
}

#define WMMA_BF16(A, B, C) \
  __builtin_amdgcn_wmma_f32_16x16x32_bf16(false, (A), false, (B), (short)0, (C), false, false)

// Stage a 32(K) x 128(N) activation panel, bf16 source, into K-transposed LDS.
__device__ __forceinline__ void stage_b_bf16(bf16* blds, const bf16* __restrict__ src,
                                             int k0, long nbase, int tid) {
  const int kl = tid >> 3;          // 0..31
  const int n0 = (tid & 7) * 16;    // 0..112
  const bf16* p = src + (long)(k0 + kl) * SP + nbase + n0;
#pragma unroll
  for (int j = 0; j < 16; ++j) blds[(n0 + j) * 32 + kl] = p[j];
}

// Same, fp32 source with fused BatchNorm (eval) -> bf16.
__device__ __forceinline__ void stage_b_f32bn(bf16* blds, const float* __restrict__ src,
                                              int k0, long nbase,
                                              const float* __restrict__ g,
                                              const float* __restrict__ b,
                                              const float* __restrict__ rm,
                                              const float* __restrict__ rv, int tid) {
  const int kl = tid >> 3;
  const int n0 = (tid & 7) * 16;
  const int c  = k0 + kl;
  const float iv = rsqrtf(rv[c] + 1e-5f);
  const float sc = g[c] * iv;
  const float sh = b[c] - rm[c] * sc;
  const float* p = src + (long)c * SP + nbase + n0;
#pragma unroll
  for (int j = 0; j < 16; ++j) blds[(n0 + j) * 32 + kl] = (bf16)(p[j] * sc + sh);
}

// ---------------------------------------------------------------------------
// K1: h = gelu(proj1(bn1(x)))        [64x64 GEMM, N=spatial]
// ---------------------------------------------------------------------------
__global__ __launch_bounds__(256) void k_pw1(
    const float* __restrict__ x, const float* __restrict__ g, const float* __restrict__ b,
    const float* __restrict__ rm, const float* __restrict__ rv,
    const float* __restrict__ w, const float* __restrict__ bias, bf16* __restrict__ h) {
  __shared__ bf16 blds[128 * 32];
  const long nbase = (long)blockIdx.x * NB;
  const int tid = threadIdx.x, lane = tid & 31, wave = tid >> 5;
  const int wm = wave & 3, wn = wave >> 2;
  const int mb = wm * 16;
  v8f acc[4];
#pragma unroll
  for (int j = 0; j < 4; ++j)
#pragma unroll
    for (int r = 0; r < 8; ++r) acc[j][r] = 0.f;

#pragma unroll
  for (int ks = 0; ks < 2; ++ks) {
    const int k0 = ks * 32;
    if (ks == 0) __builtin_prefetch(x + (long)32 * SP + nbase, 0, 1);
    __syncthreads();
    stage_b_f32bn(blds, x, k0, nbase, g, b, rm, rv, tid);
    __syncthreads();
    v16bf A = load_a_w(w, mb + (lane & 15), 64, k0, lane);
#pragma unroll
    for (int j = 0; j < 4; ++j) {
      v16bf B = load_b_lds(blds, wn * 64 + j * 16 + (lane & 15), 32, 0, lane);
      acc[j] = WMMA_BF16(A, B, acc[j]);
    }
  }
  const int mhi = (lane & 16) ? 8 : 0;
#pragma unroll
  for (int j = 0; j < 4; ++j) {
    const long n = nbase + wn * 64 + j * 16 + (lane & 15);
#pragma unroll
    for (int r = 0; r < 8; ++r) {
      const int m = mb + mhi + r;
      h[(long)m * SP + n] = (bf16)gelu_erf(acc[j][r] + bias[m]);
    }
  }
}

// ---------------------------------------------------------------------------
// K4: feat = ls1*(proj2(c3(asum)*u) + b + bn1(x)) + x   (two fused GEMMs)
// ---------------------------------------------------------------------------
__global__ __launch_bounds__(256) void k_attn_tail(
    const bf16* __restrict__ asum, const bf16* __restrict__ h, const float* __restrict__ x,
    const float* __restrict__ c3w, const float* __restrict__ c3b,
    const float* __restrict__ p2w, const float* __restrict__ p2b,
    const float* __restrict__ ls1,
    const float* __restrict__ g1, const float* __restrict__ b1,
    const float* __restrict__ rm1, const float* __restrict__ rv1,
    float* __restrict__ feat) {
  __shared__ bf16 blds[128 * 32];
  __shared__ bf16 b2[128 * 64];
  const long nbase = (long)blockIdx.x * NB;
  const int tid = threadIdx.x, lane = tid & 31, wave = tid >> 5;
  const int wm = wave & 3, wn = wave >> 2;
  const int mb = wm * 16;
  const int mhi = (lane & 16) ? 8 : 0;

  // ---- phase 1: a1 = c3 x asum ----
  v8f acc[4];
#pragma unroll
  for (int j = 0; j < 4; ++j)
#pragma unroll
    for (int r = 0; r < 8; ++r) acc[j][r] = 0.f;
#pragma unroll
  for (int ks = 0; ks < 2; ++ks) {
    const int k0 = ks * 32;
    __syncthreads();
    stage_b_bf16(blds, asum, k0, nbase, tid);
    __syncthreads();
    v16bf A = load_a_w(c3w, mb + (lane & 15), 64, k0, lane);
#pragma unroll
    for (int j = 0; j < 4; ++j) {
      v16bf B = load_b_lds(blds, wn * 64 + j * 16 + (lane & 15), 32, 0, lane);
      acc[j] = WMMA_BF16(A, B, acc[j]);
    }
  }
  // multiply by u = h, restage as phase-2 B (K=64) in LDS
#pragma unroll
  for (int j = 0; j < 4; ++j) {
    const int nl = wn * 64 + j * 16 + (lane & 15);
    const long n = nbase + nl;
#pragma unroll
    for (int r = 0; r < 8; ++r) {
      const int m = mb + mhi + r;
      const float v = (acc[j][r] + c3b[m]) * (float)h[(long)m * SP + n];
      b2[nl * 64 + m] = (bf16)v;
    }
  }
  __syncthreads();

  // ---- phase 2: proj2 x (a1*u) ----
  v8f acc2[4];
#pragma unroll
  for (int j = 0; j < 4; ++j)
#pragma unroll
    for (int r = 0; r < 8; ++r) acc2[j][r] = 0.f;
#pragma unroll
  for (int ks = 0; ks < 2; ++ks) {
    const int k0 = ks * 32;
    v16bf A = load_a_w(p2w, mb + (lane & 15), 64, k0, lane);
#pragma unroll
    for (int j = 0; j < 4; ++j) {
      v16bf B = load_b_lds(b2, wn * 64 + j * 16 + (lane & 15), 64, k0, lane);
      acc2[j] = WMMA_BF16(A, B, acc2[j]);
    }
  }
#pragma unroll
  for (int j = 0; j < 4; ++j) {
    const long n = nbase + wn * 64 + j * 16 + (lane & 15);
#pragma unroll
    for (int r = 0; r < 8; ++r) {
      const int m = mb + mhi + r;
      const float iv = rsqrtf(rv1[m] + 1e-5f);
      const float sc = g1[m] * iv;
      const float sh = b1[m] - rm1[m] * sc;
      const float xv = x[(long)m * SP + n];
      const float shortcut = xv * sc + sh;
      feat[(long)m * SP + n] = ls1[m] * (acc2[j][r] + p2b[m] + shortcut) + xv;
    }
  }
}

// ---------------------------------------------------------------------------
// K5: f1 = fc1(bn2(feat))          [256x64 GEMM]
// ---------------------------------------------------------------------------
__global__ __launch_bounds__(256) void k_fc1(
    const float* __restrict__ feat, const float* __restrict__ g, const float* __restrict__ b,
    const float* __restrict__ rm, const float* __restrict__ rv,
    const float* __restrict__ w, const float* __restrict__ bias, bf16* __restrict__ f1) {
  __shared__ bf16 blds[128 * 32];
  const long nbase = (long)blockIdx.x * NB;
  const int tid = threadIdx.x, lane = tid & 31, wave = tid >> 5;
  const int wm = wave & 3, wn = wave >> 2;
  v8f acc[4][4];
#pragma unroll
  for (int t = 0; t < 4; ++t)
#pragma unroll
    for (int j = 0; j < 4; ++j)
#pragma unroll
      for (int r = 0; r < 8; ++r) acc[t][j][r] = 0.f;

#pragma unroll
  for (int ks = 0; ks < 2; ++ks) {
    const int k0 = ks * 32;
    if (ks == 0) __builtin_prefetch(feat + (long)32 * SP + nbase, 0, 1);
    __syncthreads();
    stage_b_f32bn(blds, feat, k0, nbase, g, b, rm, rv, tid);
    __syncthreads();
#pragma unroll
    for (int t = 0; t < 4; ++t) {
      v16bf A = load_a_w(w, (wm * 4 + t) * 16 + (lane & 15), 64, k0, lane);
#pragma unroll
      for (int j = 0; j < 4; ++j) {
        v16bf B = load_b_lds(blds, wn * 64 + j * 16 + (lane & 15), 32, 0, lane);
        acc[t][j] = WMMA_BF16(A, B, acc[t][j]);
      }
    }
  }
  const int mhi = (lane & 16) ? 8 : 0;
#pragma unroll
  for (int t = 0; t < 4; ++t)
#pragma unroll
    for (int j = 0; j < 4; ++j) {
      const long n = nbase + wn * 64 + j * 16 + (lane & 15);
#pragma unroll
      for (int r = 0; r < 8; ++r) {
        const int m = (wm * 4 + t) * 16 + mhi + r;
        f1[(long)m * SP + n] = (bf16)(acc[t][j][r] + bias[m]);
      }
    }
}

// ---------------------------------------------------------------------------
// K7: out = ls2*(fc2(g)+b) + feat  [64x256 GEMM]
// ---------------------------------------------------------------------------
__global__ __launch_bounds__(256) void k_fc2(
    const bf16* __restrict__ gsrc, const float* __restrict__ w, const float* __restrict__ bias,
    const float* __restrict__ ls2, const float* __restrict__ feat, float* __restrict__ out) {
  __shared__ bf16 blds[128 * 32];
  const long nbase = (long)blockIdx.x * NB;
  const int tid = threadIdx.x, lane = tid & 31, wave = tid >> 5;
  const int wm = wave & 3, wn = wave >> 2;
  const int mb = wm * 16;
  v8f acc[4];
#pragma unroll
  for (int j = 0; j < 4; ++j)
#pragma unroll
    for (int r = 0; r < 8; ++r) acc[j][r] = 0.f;

  for (int ks = 0; ks < 8; ++ks) {
    const int k0 = ks * 32;
    if (ks < 7) __builtin_prefetch(gsrc + (long)(k0 + 32) * SP + nbase, 0, 1);
    __syncthreads();
    stage_b_bf16(blds, gsrc, k0, nbase, tid);
    __syncthreads();
    v16bf A = load_a_w(w, mb + (lane & 15), 256, k0, lane);
#pragma unroll
    for (int j = 0; j < 4; ++j) {
      v16bf B = load_b_lds(blds, wn * 64 + j * 16 + (lane & 15), 32, 0, lane);
      acc[j] = WMMA_BF16(A, B, acc[j]);
    }
  }
  const int mhi = (lane & 16) ? 8 : 0;
#pragma unroll
  for (int j = 0; j < 4; ++j) {
    const long n = nbase + wn * 64 + j * 16 + (lane & 15);
#pragma unroll
    for (int r = 0; r < 8; ++r) {
      const int m = mb + mhi + r;
      out[(long)m * SP + n] = ls2[m] * (acc[j][r] + bias[m]) + feat[(long)m * SP + n];
    }
  }
}

// ---------------------------------------------------------------------------
// Cubic depthwise conv (R=2 -> 5x5x5, R=1 -> 3x3x3), LDS-tiled.
// ---------------------------------------------------------------------------
template <int R, bool GELU, bool DUP>
__global__ __launch_bounds__(256) void k_dw_cube(
    const bf16* __restrict__ in, const float* __restrict__ w, const float* __restrict__ bias,
    bf16* __restrict__ out, bf16* __restrict__ out2) {
  constexpr int DT = 4, HT = 4, WT = 64;
  constexpr int K  = 2 * R + 1;
  constexpr int ID = DT + 2 * R, IH = HT + 2 * R, IW = WT + 2 * R;
  __shared__ bf16 tin[ID * IH * IW];
  __shared__ float tw[K * K * K];

  int bid = blockIdx.x;
  const int wx = bid % (CW / WT); bid /= (CW / WT);
  const int hy = bid % (CH / HT); bid /= (CH / HT);
  const int dz = bid % (CD / DT); bid /= (CD / DT);
  const int c  = bid;
  const int d0 = dz * DT, h0 = hy * HT, w0 = wx * WT;
  const int tid = threadIdx.x;

  if (tid < K * K * K) tw[tid] = w[c * K * K * K + tid];
  const float bv = bias[c];
  const bf16* src = in + (long)c * SP;

  for (int idx = tid; idx < ID * IH * IW; idx += 256) {
    const int iw = idx % IW; int t = idx / IW;
    const int ih = t % IH; const int id = t / IH;
    const int gd = d0 - R + id, gh = h0 - R + ih, gw = w0 - R + iw;
    bf16 v = (bf16)0.0f;
    if (gd >= 0 && gd < CD && gh >= 0 && gh < CH && gw >= 0 && gw < CW)
      v = src[(long)gd * CHW + gh * CW + gw];
    tin[idx] = v;
  }
  __syncthreads();

  for (int o = tid; o < DT * HT * WT; o += 256) {
    const int ow = o % WT; int t = o / WT;
    const int oh = t % HT; const int od = t / HT;
    float acc = bv;
#pragma unroll
    for (int kd = 0; kd < K; ++kd)
#pragma unroll
      for (int kh = 0; kh < K; ++kh)
#pragma unroll
        for (int kw = 0; kw < K; ++kw)
          acc += (float)tin[((od + kd) * IH + (oh + kh)) * IW + (ow + kw)] *
                 tw[(kd * K + kh) * K + kw];
    if (GELU) acc = gelu_erf(acc);
    const long gidx = (long)c * SP + (long)(d0 + od) * CHW + (h0 + oh) * CW + (w0 + ow);
    out[gidx] = (bf16)acc;
    if (DUP) out2[gidx] = (bf16)acc;
  }
}

// ---------------------------------------------------------------------------
// Fused separable branch: convW(2P+1) -> convH(2P+1) -> convD(3), += asum.
// Pencil tiles: full W=192, HT rows, DT depths (+halos).  The input tile is
// pulled with the CDNA5 async global->LDS engine (ASYNCcnt) when available:
// each W-row is 192*bf16 = 384 B = 24 aligned 16-B chunks, so the copy is a
// pure b128 async stream; out-of-range halo rows are zero-filled via LDS.
// ---------------------------------------------------------------------------
template <int P, int DT, int HT>
__global__ __launch_bounds__(256) void k_branch(
    const bf16* __restrict__ a,
    const float* __restrict__ wWp, const float* __restrict__ bWp,
    const float* __restrict__ wHp, const float* __restrict__ bHp,
    const float* __restrict__ wDp, const float* __restrict__ bDp,
    bf16* __restrict__ asum) {
  constexpr int KP = 2 * P + 1;
  constexpr int ID = DT + 2, IH = HT + 2 * P;
  __shared__ bf16 tin[ID * IH * CW];
  __shared__ bf16 t1[ID * IH * CW];
  __shared__ bf16 t2[ID * HT * CW];
  __shared__ float wW[KP], wH[KP], wD[3];

  int bid = blockIdx.x;
  const int hy = bid % (CH / HT); bid /= (CH / HT);
  const int dz = bid % (CD / DT); bid /= (CD / DT);
  const int c  = bid;
  const int d0 = dz * DT, h0 = hy * HT;
  const int tid = threadIdx.x;

  if (tid < KP) { wW[tid] = wWp[c * KP + tid]; wH[tid] = wHp[c * KP + tid]; }
  if (tid < 3)  wD[tid] = wDp[c * 3 + tid];
  const float bW = bWp[c], bH = bHp[c], bD = bDp[c];
  const bf16* src = a + (long)c * SP;

#if defined(HAVE_ASYNC_LDS)
  // async tile load: 24 x b128 chunks per row
  for (int rc = tid; rc < ID * IH * 24; rc += 256) {
    const int chunk = rc % 24; const int t = rc / 24;
    const int ih = t % IH; const int id = t / IH;
    const int gd = d0 - 1 + id, gh = h0 - P + ih;
    bf16* ldst = tin + t * CW + chunk * 8;
    if (gd >= 0 && gd < CD && gh >= 0 && gh < CH) {
      const bf16* gsrc = src + (long)gd * CHW + gh * CW + chunk * 8;
      __builtin_amdgcn_global_load_async_to_lds_b128(to_gm128(gsrc), to_lds128(ldst), 0, 0);
    } else {
      *(uint4*)ldst = make_uint4(0u, 0u, 0u, 0u);
    }
  }
  wait_async0();
  __syncthreads();
#else
  for (int idx = tid; idx < ID * IH * CW; idx += 256) {
    const int w = idx % CW; int t = idx / CW;
    const int ih = t % IH; const int id = t / IH;
    const int gd = d0 - 1 + id, gh = h0 - P + ih;
    bf16 v = (bf16)0.0f;
    if (gd >= 0 && gd < CD && gh >= 0 && gh < CH) v = src[(long)gd * CHW + gh * CW + w];
    tin[idx] = v;
  }
  __syncthreads();
#endif

  // conv along W (zero rows that are outside the valid (d,h) domain)
  for (int idx = tid; idx < ID * IH * CW; idx += 256) {
    const int w = idx % CW; const int t = idx / CW;
    const int ih = t % IH; const int id = t / IH;
    const int gd = d0 - 1 + id, gh = h0 - P + ih;
    float acc = 0.f;
    if (gd >= 0 && gd < CD && gh >= 0 && gh < CH) {
      acc = bW;
#pragma unroll
      for (int i = 0; i < KP; ++i) {
        const int ww = w - P + i;
        if (ww >= 0 && ww < CW) acc += (float)tin[t * CW + ww] * wW[i];
      }
    }
    t1[idx] = (bf16)acc;
  }
  __syncthreads();

  // conv along H (zero rows at invalid depth so convD sees true zero pad)
  for (int idx = tid; idx < ID * HT * CW; idx += 256) {
    const int w = idx % CW; const int t = idx / CW;
    const int oh = t % HT; const int id = t / HT;
    const int gd = d0 - 1 + id;
    float acc = 0.f;
    if (gd >= 0 && gd < CD) {
      acc = bH;
#pragma unroll
      for (int i = 0; i < KP; ++i) acc += (float)t1[(id * IH + oh + i) * CW + w] * wH[i];
    }
    t2[idx] = (bf16)acc;
  }
  __syncthreads();

  // conv along D + accumulate into asum
  for (int idx = tid; idx < DT * HT * CW; idx += 256) {
    const int w = idx % CW; const int t = idx / CW;
    const int oh = t % HT; const int od = t / HT;
    float acc = bD;
#pragma unroll
    for (int i = 0; i < 3; ++i) acc += (float)t2[((od + i) * HT + oh) * CW + w] * wD[i];
    const long gidx = (long)c * SP + (long)(d0 + od) * CHW + (h0 + oh) * CW + w;
    asum[gidx] = (bf16)((float)asum[gidx] + acc);
  }
}

// ---------------------------------------------------------------------------
// Host launch
// ---------------------------------------------------------------------------
enum {
  IN_X = 0, IN_BN1_G, IN_BN1_B, IN_BN1_RM, IN_BN1_RV,
  IN_P1_W, IN_P1_B, IN_C0_W, IN_C0_B,
  IN_C01_W, IN_C01_B, IN_C02_W, IN_C02_B, IN_C03_W, IN_C03_B,
  IN_C11_W, IN_C11_B, IN_C12_W, IN_C12_B, IN_C13_W, IN_C13_B,
  IN_C21_W, IN_C21_B, IN_C22_W, IN_C22_B, IN_C23_W, IN_C23_B,
  IN_C3_W, IN_C3_B, IN_P2_W, IN_P2_B,
  IN_BN2_G, IN_BN2_B, IN_BN2_RM, IN_BN2_RV,
  IN_FC1_W, IN_FC1_B, IN_MDW_W, IN_MDW_B, IN_FC2_W, IN_FC2_B,
  IN_LS1, IN_LS2
};

extern "C" void kernel_launch(void* const* d_in, const int* in_sizes, int n_in,
                              void* d_out, int out_size, void* d_ws, size_t ws_size,
                              hipStream_t stream) {
  auto F = [&](int i) { return (const float*)d_in[i]; };
  char* ws = (char*)d_ws;

  // scratch layout (liveness-based reuse):
  float* feat = (float*)(ws + 0);                 // 113.2 MB fp32, live to end
  bf16*  h    = (bf16*)(ws + 113246208ull);       // 56.6 MB, dead after attn_tail
  bf16*  aob  = (bf16*)(ws + 169869312ull);       // 56.6 MB (c0 output), dead after attn_tail
  bf16*  asum = (bf16*)(ws + 226492416ull);       // 56.6 MB, dead after attn_tail
  bf16*  f1   = (bf16*)(ws + 113246208ull);       // 226.5 MB, reuses h/a/asum region
  bf16*  gbuf = (bf16*)(ws + 339738624ull);       // 226.5 MB
  (void)ws_size; (void)n_in; (void)in_sizes; (void)out_size;

  const dim3 blk(256);

  // 1) h = gelu(proj1(bn1(x)))
  k_pw1<<<NBLK, blk, 0, stream>>>(F(IN_X), F(IN_BN1_G), F(IN_BN1_B), F(IN_BN1_RM),
                                  F(IN_BN1_RV), F(IN_P1_W), F(IN_P1_B), h);
  // 2) a = dw5x5x5(h); asum = a
  k_dw_cube<2, false, true><<<DIM * 432, blk, 0, stream>>>(h, F(IN_C0_W), F(IN_C0_B), aob, asum);
  // 3-5) separable branches accumulate into asum
  k_branch<3, 2, 8><<<DIM * (CD / 2) * (CH / 8), blk, 0, stream>>>(
      aob, F(IN_C01_W), F(IN_C01_B), F(IN_C02_W), F(IN_C02_B), F(IN_C03_W), F(IN_C03_B), asum);
  k_branch<5, 2, 6><<<DIM * (CD / 2) * (CH / 6), blk, 0, stream>>>(
      aob, F(IN_C11_W), F(IN_C11_B), F(IN_C12_W), F(IN_C12_B), F(IN_C13_W), F(IN_C13_B), asum);
  k_branch<10, 1, 4><<<DIM * CD * (CH / 4), blk, 0, stream>>>(
      aob, F(IN_C21_W), F(IN_C21_B), F(IN_C22_W), F(IN_C22_B), F(IN_C23_W), F(IN_C23_B), asum);
  // 6) feat = ls1*(proj2(c3(asum)*h) + bn1(x)) + x
  k_attn_tail<<<NBLK, blk, 0, stream>>>(asum, h, F(IN_X), F(IN_C3_W), F(IN_C3_B),
                                        F(IN_P2_W), F(IN_P2_B), F(IN_LS1),
                                        F(IN_BN1_G), F(IN_BN1_B), F(IN_BN1_RM), F(IN_BN1_RV),
                                        feat);
  // 7) f1 = fc1(bn2(feat))
  k_fc1<<<NBLK, blk, 0, stream>>>(feat, F(IN_BN2_G), F(IN_BN2_B), F(IN_BN2_RM), F(IN_BN2_RV),
                                  F(IN_FC1_W), F(IN_FC1_B), f1);
  // 8) g = gelu(dw3x3x3(f1))
  k_dw_cube<1, true, false><<<HIDC * 432, blk, 0, stream>>>(f1, F(IN_MDW_W), F(IN_MDW_B),
                                                            gbuf, (bf16*)nullptr);
  // 9) out = ls2*(fc2(g)+b) + feat
  k_fc2<<<NBLK, blk, 0, stream>>>(gbuf, F(IN_FC2_W), F(IN_FC2_B), F(IN_LS2), feat,
                                  (float*)d_out);
}
